// MMoE_31825707663784
// MI455X (gfx1250) — matrix operand order repaired and checked by reading, not compiled, not run on hardware
//
#include <hip/hip_runtime.h>
#include <hip/hip_bf16.h>

#define B_ 16384
#define F_ 128
#define H_ 256
#define E_ 8
#define T_ 3

typedef __attribute__((ext_vector_type(16))) __bf16 v16bf;
typedef __attribute__((ext_vector_type(8)))  __bf16 v8bf;
typedef __attribute__((ext_vector_type(4)))  __bf16 v4bf;
typedef __attribute__((ext_vector_type(8)))  float  v8f;

static __device__ __forceinline__ v8f wmma_bf16(v16bf a, v16bf b, v8f c) {
    // D = A(16x32 bf16) * B(32x16 bf16) + C(16x16 f32)
    return __builtin_amdgcn_wmma_f32_16x16x32_bf16(false, a, false, b, (short)0, c, false, false);
}

// A fragment: row-major bf16 [.., K], 16 rows at row0, K window [kb, kb+32)
// lane<16: row=lane, K=kb+0..15 (elems 0..15)
// lane>=16: row=lane-16, K=kb+16..31
static __device__ __forceinline__ v16bf load_a_frag(const __bf16* base, int ld, int row0, int kb, int lane) {
    int r = lane & 15, half = lane >> 4;
    const __bf16* p = base + (size_t)(row0 + r) * ld + kb + 16 * half;
    v16bf a;
    ((v8bf*)&a)[0] = *(const v8bf*)(p);
    ((v8bf*)&a)[1] = *(const v8bf*)(p + 8);
    return a;
}

// B fragment: W row-major [N, K] (reduction dim contiguous), cols [n0, n0+16), K window [kb, kb+32)
// lane<16: col=lane, K=kb+0..15 ; lane>=16: col=lane-16, K=kb+16..31 (contiguous 32B load)
static __device__ __forceinline__ v16bf load_b_frag(const __bf16* W, int ld, int n0, int kb, int lane) {
    int c = lane & 15, half = lane >> 4;
    return *(const v16bf*)(W + (size_t)(n0 + c) * ld + kb + 16 * half);
}

// ---------------- weight f32 -> bf16 (vectorized x4) ----------------
__global__ void cvt_f32_bf16(const float* __restrict__ src, __bf16* __restrict__ dst, int n4) {
    int i = blockIdx.x * blockDim.x + threadIdx.x;
    if (i < n4) {
        float4 v = ((const float4*)src)[i];
        v4bf o;
        o[0] = (__bf16)v.x; o[1] = (__bf16)v.y; o[2] = (__bf16)v.z; o[3] = (__bf16)v.w;
        ((v4bf*)dst)[i] = o;
    }
}

// ---------------- embedding gather (+bf16) + gate logits + softmax ----------------
// block = 256 threads = 8 waves; wave w handles row b = blockIdx.x*8 + w
__global__ void gather_gates(const int* __restrict__ user,
                             const float* __restrict__ emb_table,
                             const float* __restrict__ gate_w,   // [T,E,F]
                             const float* __restrict__ gate_b,   // [T,E]
                             __bf16* __restrict__ emb_bf,        // [B,F]
                             float* __restrict__ gates) {        // [B,T,E] softmaxed
    __shared__ __align__(16) float semb[8][F_];
    int wave = threadIdx.x >> 5, lane = threadIdx.x & 31;
    int b = blockIdx.x * 8 + wave;
    int u = user[b];
    float4 v = ((const float4*)(emb_table + (size_t)u * F_))[lane];
    *(float4*)&semb[wave][lane * 4] = v;
    v4bf o;
    o[0] = (__bf16)v.x; o[1] = (__bf16)v.y; o[2] = (__bf16)v.z; o[3] = (__bf16)v.w;
    *(v4bf*)(emb_bf + (size_t)b * F_ + lane * 4) = o;
    __syncthreads();
    if (lane < T_ * E_) {
        const float* w = gate_w + lane * F_;   // lane = t*E + e
        float acc = gate_b[lane];
        #pragma unroll 4
        for (int f = 0; f < F_; f += 4) {
            float4 wv = *(const float4*)(w + f);
            float4 ev = *(const float4*)(&semb[wave][f]);
            acc += ev.x * wv.x + ev.y * wv.y + ev.z * wv.z + ev.w * wv.w;
        }
        // softmax over E=8 within xor-groups of 8 lanes
        float m = acc;
        for (int off = 1; off < 8; off <<= 1) m = fmaxf(m, __shfl_xor(m, off, 32));
        float ex = __expf(acc - m);
        float s = ex;
        for (int off = 1; off < 8; off <<= 1) s += __shfl_xor(s, off, 32);
        gates[(size_t)b * (T_ * E_) + lane] = ex / s;
    }
}

// ---------------- experts GEMM + bias + relu + gate-weighted mixture ----------------
// grid = 2048 blocks of 256 thr (8 waves). rowblk = bid>>1 (16 rows), wave covers one 16-wide h tile.
// A fragments (invariant across experts) hoisted; expert loop unrolled x2 to bound live B/C tiles.
__global__ void experts_mix(const __bf16* __restrict__ emb,      // [B,F] bf16
                            const __bf16* __restrict__ wexp,     // [E*H, F] bf16
                            const float*  __restrict__ expert_b, // [E,H]
                            const float*  __restrict__ gates,    // [B,T,E]
                            __bf16* __restrict__ mix) {          // [T,B,H] bf16
    __shared__ float sgates[16 * T_ * E_];
    int rowblk = blockIdx.x >> 1;
    int hblk   = ((blockIdx.x & 1) << 3) + (threadIdx.x >> 5);   // 0..15
    int lane   = threadIdx.x & 31;
    int row0   = rowblk * 16;
    for (int i = threadIdx.x; i < 16 * T_ * E_; i += 256)
        sgates[i] = gates[(size_t)row0 * (T_ * E_) + i];
    __syncthreads();

    int half = lane >> 4, c = lane & 15;
    int hcol = hblk * 16 + c;

    // A tile (16 rows x K=128) loaded once, reused by all 8 experts: 4 frags = 32 VGPRs
    v16bf afrag[4];
    #pragma unroll
    for (int k4 = 0; k4 < 4; ++k4)
        afrag[k4] = load_a_frag(emb, F_, row0, k4 * 32, lane);

    v8f acc0 = {}, acc1 = {}, acc2 = {};
    #pragma unroll 2
    for (int e = 0; e < E_; ++e) {
        v8f cacc = {};
        const __bf16* We = wexp + (size_t)e * H_ * F_;
        #pragma unroll
        for (int k4 = 0; k4 < 4; ++k4) {
            v16bf bm = load_b_frag(We, F_, hblk * 16, k4 * 32, lane);
            cacc = wmma_bf16(afrag[k4], bm, cacc);
        }
        float bias = expert_b[e * H_ + hcol];
        #pragma unroll
        for (int i = 0; i < 8; ++i) {
            float v = cacc[i] + bias;
            v = v > 0.f ? v : 0.f;               // relu
            int rr = i + 8 * half;
            acc0[i] += sgates[rr * (T_ * E_) + 0 * E_ + e] * v;
            acc1[i] += sgates[rr * (T_ * E_) + 1 * E_ + e] * v;
            acc2[i] += sgates[rr * (T_ * E_) + 2 * E_ + e] * v;
        }
    }
    #pragma unroll
    for (int i = 0; i < 8; ++i) {
        size_t row = (size_t)(row0 + i + 8 * half);
        mix[(0 * (size_t)B_ + row) * H_ + hcol] = (__bf16)acc0[i];
        mix[(1 * (size_t)B_ + row) * H_ + hcol] = (__bf16)acc1[i];
        mix[(2 * (size_t)B_ + row) * H_ + hcol] = (__bf16)acc2[i];
    }
}

// ---------------- tower hidden layer: Y[t] = relu(X[t] * W[t]^T + b[t]) ----------------
// grid = (1024, 3); block 256 = 8 waves; each wave: rows [bx*16,+16), two 16-col tiles (A reused)
__global__ void tower_gemm(const __bf16* __restrict__ X,   // [T,B,H] bf16
                           const __bf16* __restrict__ W,   // [T,H,H] bf16 (out-row, in-col)
                           const float*  __restrict__ bias,// [T,H]
                           __bf16* __restrict__ Y) {       // [T,B,H] bf16
    int t = blockIdx.y;
    int row0 = blockIdx.x * 16;
    int lane = threadIdx.x & 31;
    int h0 = (threadIdx.x >> 5) * 32;
    int half = lane >> 4, c = lane & 15;
    const __bf16* Xt = X + (size_t)t * B_ * H_;
    const __bf16* Wt = W + (size_t)t * H_ * H_;
    v8f acc0 = {}, acc1 = {};
    #pragma unroll 4
    for (int kb = 0; kb < H_; kb += 32) {
        v16bf a  = load_a_frag(Xt, H_, row0, kb, lane);
        v16bf b0 = load_b_frag(Wt, H_, h0, kb, lane);
        v16bf b1 = load_b_frag(Wt, H_, h0 + 16, kb, lane);
        acc0 = wmma_bf16(a, b0, acc0);
        acc1 = wmma_bf16(a, b1, acc1);
    }
    float bias0 = bias[t * H_ + h0 + c];
    float bias1 = bias[t * H_ + h0 + 16 + c];
    #pragma unroll
    for (int i = 0; i < 8; ++i) {
        size_t row = (size_t)(row0 + i + 8 * half);
        float v0 = acc0[i] + bias0; v0 = v0 > 0.f ? v0 : 0.f;
        float v1 = acc1[i] + bias1; v1 = v1 > 0.f ? v1 : 0.f;
        Y[((size_t)t * B_ + row) * H_ + h0 + c]      = (__bf16)v0;
        Y[((size_t)t * B_ + row) * H_ + h0 + 16 + c] = (__bf16)v1;
    }
}

// ---------------- final GEMV: out[b,t] = dot(h2[t,b,:], w3[t,:]) + b3[t] ----------------
// one wave per (t,b): lane loads 8 contiguous bf16, shuffle reduce
__global__ void tower_out(const __bf16* __restrict__ h2, // [T,B,H]
                          const float*  __restrict__ w3, // [T,1,H]
                          const float*  __restrict__ b3, // [T,1]
                          float* __restrict__ out) {     // [B,T]
    int wave = threadIdx.x >> 5, lane = threadIdx.x & 31;
    int idx = blockIdx.x * 8 + wave;        // idx = t*B + b
    int t = idx >> 14;                      // B_ = 1<<14
    int b = idx & (B_ - 1);
    const __bf16* p = h2 + (size_t)idx * H_ + lane * 8;
    const float*  w = w3 + t * H_ + lane * 8;
    float acc = 0.f;
    #pragma unroll
    for (int j = 0; j < 8; ++j) acc += (float)p[j] * w[j];
    for (int off = 16; off; off >>= 1) acc += __shfl_xor(acc, off, 32);
    if (lane == 0) out[(size_t)b * T_ + t] = acc + b3[t];
}

extern "C" void kernel_launch(void* const* d_in, const int* in_sizes, int n_in,
                              void* d_out, int out_size, void* d_ws, size_t ws_size,
                              hipStream_t stream) {
    (void)in_sizes; (void)n_in; (void)out_size; (void)ws_size;
    const int*   user      = (const int*)  d_in[0];
    const float* emb_table = (const float*)d_in[1];
    const float* expert_w  = (const float*)d_in[2];
    const float* expert_b  = (const float*)d_in[3];
    const float* gate_w    = (const float*)d_in[4];
    const float* gate_b    = (const float*)d_in[5];
    const float* tower_w1  = (const float*)d_in[6];
    const float* tower_b1  = (const float*)d_in[7];
    const float* tower_w2  = (const float*)d_in[8];
    const float* tower_b2  = (const float*)d_in[9];
    const float* tower_w3  = (const float*)d_in[10];
    const float* tower_b3  = (const float*)d_in[11];
    float* out = (float*)d_out;

    // workspace layout (bytes)
    char* ws = (char*)d_ws;
    __bf16* emb_bf = (__bf16*)(ws);                          //  4 MB  [B,F]
    float*  gates  = (float*) (ws + 4194304);                //  1.5MB [B,T,E]
    __bf16* wexp_bf= (__bf16*)(ws + 5767168);                //  0.5MB [E*H,F]
    __bf16* w1_bf  = (__bf16*)(ws + 6291456);                //  384KB [T,H,H]
    __bf16* w2_bf  = (__bf16*)(ws + 6684672);                //  384KB [T,H,H]
    __bf16* mix    = (__bf16*)(ws + 7077888);                //  24 MB [T,B,H]
    __bf16* h1     = (__bf16*)(ws + 32243712);               //  24 MB [T,B,H]
    __bf16* h2     = (__bf16*)(ws + 57409536);               //  24 MB [T,B,H]

    // 1) weight conversion (tiny, L2-resident thereafter)
    cvt_f32_bf16<<<(E_*H_*F_/4 + 255)/256, 256, 0, stream>>>(expert_w, wexp_bf, E_*H_*F_/4);
    cvt_f32_bf16<<<(T_*H_*H_/4 + 255)/256, 256, 0, stream>>>(tower_w1, w1_bf, T_*H_*H_/4);
    cvt_f32_bf16<<<(T_*H_*H_/4 + 255)/256, 256, 0, stream>>>(tower_w2, w2_bf, T_*H_*H_/4);

    // 2) gather + gates(+softmax)
    gather_gates<<<B_/8, 256, 0, stream>>>(user, emb_table, gate_w, gate_b, emb_bf, gates);

    // 3) experts GEMM + relu + mixture (fused; experts tensor never materialized)
    experts_mix<<<(B_/16)*2, 256, 0, stream>>>(emb_bf, wexp_bf, expert_b, gates, mix);

    // 4) towers
    tower_gemm<<<dim3(B_/16, T_), 256, 0, stream>>>(mix, w1_bf, tower_b1, h1);
    tower_gemm<<<dim3(B_/16, T_), 256, 0, stream>>>(h1,  w2_bf, tower_b2, h2);

    // 5) final GEMV
    tower_out<<<B_*T_/8, 256, 0, stream>>>(h2, tower_w3, tower_b3, out);
}
